// MyModel_87522843558583
// MI455X (gfx1250) — matrix operand order repaired and checked by reading, not compiled, not run on hardware
//
#include <hip/hip_runtime.h>

// ---------------- problem constants ----------------
static constexpr int EMBD  = 32;
static constexpr int HIDD  = 64;
static constexpr int SEQL  = 400;
static constexpr int LABN  = 10;
static constexpr int BATCH = 512;
static constexpr int G4H   = 4 * HIDD;   // 256

// ---------------- CDNA5 WMMA types ----------------
typedef __attribute__((ext_vector_type(16))) __bf16 v16bf;
typedef __attribute__((ext_vector_type(8)))  float  v8f;

union FragU {
    unsigned short s[16];
    uint4          u[2];
    v16bf          v;
};
union Q8 {
    uint4          u;
    unsigned short s[8];
};

__device__ __forceinline__ unsigned short f2bf(float f) {
    unsigned u = __builtin_bit_cast(unsigned, f);
    u += 0x7FFFu + ((u >> 16) & 1u);          // round-to-nearest-even
    return (unsigned short)(u >> 16);
}
__device__ __forceinline__ float bf2f(unsigned short h) {
    return __builtin_bit_cast(float, ((unsigned)h) << 16);
}
__device__ __forceinline__ float fast_sig(float x) {
    float e = __builtin_amdgcn_exp2f(-1.4426950408889634f * x);
    return __builtin_amdgcn_rcpf(1.0f + e);
}
__device__ __forceinline__ float fast_tanh(float x) {
    float e = __builtin_amdgcn_exp2f(-2.8853900817779268f * x);
    return __builtin_amdgcn_rcpf(1.0f + e) * 2.0f - 1.0f;
}

// ---- CDNA5 split workgroup barrier: signal early, wait late ----
// s_wait_dscnt 0 before the signal guarantees this wave's LDS reads/writes
// have completed (LDS is WGP-coherent once DScnt drains).
__device__ __forceinline__ void bar_signal_lds() {
    asm volatile("s_wait_dscnt 0\n\ts_barrier_signal -1" ::: "memory");
}
__device__ __forceinline__ void bar_wait() {
    asm volatile("s_barrier_wait -1" ::: "memory");
}

// ------------------------------------------------------------------
// Kernel 1: enhanced embedding -> bf16, layout [SEQ][BATCH][32]
// emb[t][b][k] = 0.5*mean_k(table[id_tok]) + 0.5*table[cur_tok][k]
// ------------------------------------------------------------------
__global__ void emb_kernel(const int* __restrict__ ids,
                           const float* __restrict__ table,
                           unsigned short* __restrict__ emb)
{
    int tok  = blockIdx.x * (blockDim.x >> 5) + (threadIdx.x >> 5);
    int lane = threadIdx.x & 31;
    if (tok >= BATCH * SEQL) return;
    int b = tok / SEQL;
    int t = tok - b * SEQL;
    long base = ((long)b * SEQL + t) * 3;
    int id_tok  = ids[base + 0];
    int cur_tok = ids[base + 2];
    float idv  = table[(long)id_tok  * EMBD + lane];
    float curv = table[(long)cur_tok * EMBD + lane];
    float s = idv;
    #pragma unroll
    for (int off = 16; off; off >>= 1) s += __shfl_xor(s, off, 32);
    float mean_id = s * (1.0f / 32.0f);
    emb[((long)t * BATCH + b) * EMBD + lane] = f2bf(0.5f * mean_id + 0.5f * curv);
}

// ------------------------------------------------------------------
// Kernel 2: persistent bidirectional LSTM.
// grid = 64 blocks: blockIdx&1 = direction, blockIdx>>1 = 16-row batch tile.
// 4 waves/block: wave w owns hidden columns [16w,16w+16) -> its i/f/g/o tiles.
// Per step per wave: 12 x v_wmma_f32_16x16x32_bf16, split-barrier pipelined.
// ------------------------------------------------------------------
__global__ void __launch_bounds__(128)
lstm_kernel(const unsigned short* __restrict__ emb,   // [SEQ][BATCH][32] bf16
            const float* __restrict__ Wk_f, const float* __restrict__ Wr_f, const float* __restrict__ b_f,
            const float* __restrict__ Wk_b, const float* __restrict__ Wr_b, const float* __restrict__ b_b,
            float* __restrict__ maxf,                 // [SEQ][BATCH]
            float* __restrict__ maxb)                 // [SEQ][BATCH]
{
    __shared__ __align__(16) unsigned short h_lds[16 * HIDD];  // h tile, bf16 row-major

    const int dir  = blockIdx.x & 1;
    const int b0   = (blockIdx.x >> 1) * 16;
    const int wave = threadIdx.x >> 5;     // 0..3 hidden tile
    const int lane = threadIdx.x & 31;
    const int lm   = lane & 15;
    const int hi   = lane >> 4;            // lane half (A/B layout selector)

    const float* Wk = dir ? Wk_b : Wk_f;
    const float* Wr = dir ? Wr_b : Wr_f;
    const float* bb = dir ? b_b  : b_f;
    float* maxout   = dir ? maxb : maxf;

    // ---- build B fragments (resident for all 400 steps) ----
    // B layout (16-bit KxN): lane<16 -> col lm, K 0..7,16..23; lane>=16 -> col lm, K 8..15,24..31.
    v16bf WkB[4];          // [gate], K=32
    v16bf WrB[4][2];       // [gate][k-chunk], K=64
    float bias[4];
    #pragma unroll
    for (int g = 0; g < 4; ++g) {
        const int col = g * HIDD + wave * 16 + lm;
        bias[g] = bb[col];
        FragU fk;
        #pragma unroll
        for (int e = 0; e < 16; ++e)
            fk.s[e] = f2bf(Wk[(hi * 8 + (e & 7) + (e >> 3) * 16) * G4H + col]);
        WkB[g] = fk.v;
        #pragma unroll
        for (int ch = 0; ch < 2; ++ch) {
            FragU fr;
            #pragma unroll
            for (int e = 0; e < 16; ++e)
                fr.s[e] = f2bf(Wr[(ch * 32 + hi * 8 + (e & 7) + (e >> 3) * 16) * G4H + col]);
            WrB[g][ch] = fr.v;
        }
    }

    for (int i = threadIdx.x; i < 16 * HIDD; i += blockDim.x) h_lds[i] = 0;
    v8f c = {};

    const int k0   = hi * 8;               // A layout: K start for this lane half
    const int arow = b0 + lm;              // A row = batch row

    // preload x(t=0) A-fragment while h is being zeroed
    FragU xa;
    {
        const int tt0 = dir ? (SEQL - 1) : 0;
        const unsigned short* p = emb + (((long)tt0 * BATCH + arow) * EMBD + k0);
        xa.u[0] = *(const uint4*)(p);
        xa.u[1] = *(const uint4*)(p + 16);
    }
    __syncthreads();                       // h_lds zero-init visible

    for (int t = 0; t < SEQL; ++t) {
        const int tt = dir ? (SEQL - 1 - t) : t;

        // ---- read h(t-1) A-fragments from LDS, then signal "reads done" ----
        FragU ha0, ha1;
        {
            const unsigned short* p = h_lds + (lm * HIDD + k0);
            ha0.u[0] = *(const uint4*)(p);
            ha0.u[1] = *(const uint4*)(p + 16);
            ha1.u[0] = *(const uint4*)(p + 32);
            ha1.u[1] = *(const uint4*)(p + 48);
        }
        bar_signal_lds();                  // overlap barrier with WMMA + gates

        // ---- z = b + x*Wk + h*Wr for this wave's i/f/g/o tiles ----
        v8f acc[4];
        #pragma unroll
        for (int g = 0; g < 4; ++g) {
            v8f a;
            #pragma unroll
            for (int j = 0; j < 8; ++j) a[j] = bias[g];
            a = __builtin_amdgcn_wmma_f32_16x16x32_bf16(false, xa.v,  false, WkB[g],    (short)0, a, false, false);
            a = __builtin_amdgcn_wmma_f32_16x16x32_bf16(false, ha0.v, false, WrB[g][0], (short)0, a, false, false);
            a = __builtin_amdgcn_wmma_f32_16x16x32_bf16(false, ha1.v, false, WrB[g][1], (short)0, a, false, false);
            acc[g] = a;
        }

        // gates (Keras order i,f,g,o), state update in f32 registers
        float hv[8];
        #pragma unroll
        for (int j = 0; j < 8; ++j) {
            float ig = fast_sig(acc[0][j]);
            float fg = fast_sig(acc[1][j]);
            float gg = fast_tanh(acc[2][j]);
            float og = fast_sig(acc[3][j]);
            float cn = fg * c[j] + ig * gg;
            c[j]  = cn;
            hv[j] = og * fast_tanh(cn);
        }

        bar_wait();                        // all waves' h(t-1) reads complete

        // ---- write h(t) tile (C layout -> row-major bf16 LDS) ----
        {
            unsigned short* p = h_lds + (hi * 8) * HIDD + wave * 16 + lm;
            #pragma unroll
            for (int j = 0; j < 8; ++j) p[j * HIDD] = f2bf(hv[j]);
        }
        bar_signal_lds();                  // overlap barrier with next x load

        // ---- software-pipelined load of x(t+1) + prefetch x(t+2) ----
        if (t + 1 < SEQL) {
            const int tn = dir ? (SEQL - 2 - t) : (t + 1);
            const unsigned short* p = emb + (((long)tn * BATCH + arow) * EMBD + k0);
            xa.u[0] = *(const uint4*)(p);
            xa.u[1] = *(const uint4*)(p + 16);
            if (t + 2 < SEQL) {
                const int tp = dir ? (SEQL - 3 - t) : (t + 2);
                __builtin_prefetch(emb + (((long)tp * BATCH + arow) * EMBD + k0), 0, 3);
            }
        }

        bar_wait();                        // h(t) visible to all waves

        // wave 0: per-batch-row max over all 64 hidden units of this direction
        if (wave == 0) {
            const unsigned short* p = h_lds + lm * HIDD + hi * 32;
            float m = -3.0e38f;
            #pragma unroll
            for (int q = 0; q < 4; ++q) {
                Q8 d; d.u = *(const uint4*)(p + q * 8);
                #pragma unroll
                for (int e = 0; e < 8; ++e) m = fmaxf(m, bf2f(d.s[e]));
            }
            m = fmaxf(m, __shfl_xor(m, 16, 32));
            if (hi == 0) maxout[(long)tt * BATCH + b0 + lm] = m;
        }
    }
}

// ------------------------------------------------------------------
// Kernel 3: pooled = max(maxf, maxb); out = pooled @ Wd + bd
// ------------------------------------------------------------------
__global__ void head_kernel(const float* __restrict__ maxf,
                            const float* __restrict__ maxb,
                            const float* __restrict__ Wd,
                            const float* __restrict__ bd,
                            float* __restrict__ out)
{
    int b = blockIdx.x * blockDim.x + threadIdx.x;
    if (b >= BATCH) return;
    float acc[LABN];
    #pragma unroll
    for (int l = 0; l < LABN; ++l) acc[l] = bd[l];
    for (int t = 0; t < SEQL; ++t) {
        float m = fmaxf(maxf[(long)t * BATCH + b], maxb[(long)t * BATCH + b]);
        #pragma unroll
        for (int l = 0; l < LABN; ++l) acc[l] += m * Wd[t * LABN + l];
    }
    #pragma unroll
    for (int l = 0; l < LABN; ++l) out[b * LABN + l] = acc[l];
}

// ------------------------------------------------------------------
extern "C" void kernel_launch(void* const* d_in, const int* in_sizes, int n_in,
                              void* d_out, int out_size, void* d_ws, size_t ws_size,
                              hipStream_t stream)
{
    const int*   ids  = (const int*)  d_in[0];
    const float* tab  = (const float*)d_in[1];
    const float* Wk_f = (const float*)d_in[2];
    const float* Wr_f = (const float*)d_in[3];
    const float* b_f  = (const float*)d_in[4];
    const float* Wk_b = (const float*)d_in[5];
    const float* Wr_b = (const float*)d_in[6];
    const float* b_b  = (const float*)d_in[7];
    const float* Wd   = (const float*)d_in[8];
    const float* bd   = (const float*)d_in[9];
    float* out = (float*)d_out;

    char* ws = (char*)d_ws;
    unsigned short* emb = (unsigned short*)ws;                          // SEQ*B*32 bf16 (~12.5 MB)
    size_t embBytes = (size_t)SEQL * BATCH * EMBD * sizeof(unsigned short);
    float* maxf = (float*)(ws + embBytes);                              // SEQ*B f32
    float* maxb = maxf + (size_t)SEQL * BATCH;                          // SEQ*B f32

    {   // embeddings: one wave per token
        int tokens = BATCH * SEQL;
        int tpb = 256, perBlk = tpb / 32;
        emb_kernel<<<(tokens + perBlk - 1) / perBlk, tpb, 0, stream>>>(ids, tab, emb);
    }
    // persistent bi-LSTM: 32 batch tiles x 2 directions, 4 waves each
    lstm_kernel<<<(BATCH / 16) * 2, 128, 0, stream>>>(emb, Wk_f, Wr_f, b_f,
                                                      Wk_b, Wr_b, b_b, maxf, maxb);
    // dense head
    head_kernel<<<(BATCH + 255) / 256, 256, 0, stream>>>(maxf, maxb, Wd, bd, out);
}
